// GroupedTopKRouter_74517682585987
// MI455X (gfx1250) — matrix operand order repaired and checked by reading, not compiled, not run on hardware
//
#include <hip/hip_runtime.h>
#include <hip/hip_bf16.h>

// ---------------------------------------------------------------------------
// Grouped top-k MoE router for MI455X (gfx1250, wave32, WMMA).
//   probs = softmax(x @ W^T); grouped top-2 -> top-4 groups -> top-8 experts
//   + aux load-balancing loss.
// GEMM: v_wmma_f32_16x16x32_bf16, software-pipelined, bf16 staged in LDS.
// ---------------------------------------------------------------------------

typedef __attribute__((ext_vector_type(16))) __bf16        v16bf;
typedef __attribute__((ext_vector_type(8)))  float         v8f;
typedef __attribute__((ext_vector_type(8)))  unsigned int  v8u;

#define T_TOK   16384
#define D_DIM   4096
#define E_EXP   256
#define G_GRP   8
#define EPG     32
#define TOPK    8
#define NSEL    4

#define MB      64                    // tokens per block
#define KC      32                    // k-chunk (matches WMMA K)
#define NKC     (D_DIM / KC)          // 128 chunks
#define AP_STR  20                    // packed-A LDS row stride (dwords, 80B)
#define LOG2E   1.4426950408889634f

// float -> bf16 (round to nearest even), packed pair into one dword
__device__ __forceinline__ unsigned int f2bf(float f) {
  unsigned int u = __float_as_uint(f);
  return (u + 0x7FFFu + ((u >> 16) & 1u)) >> 16;
}
__device__ __forceinline__ unsigned int pack2(float lo, float hi) {
  return f2bf(lo) | (f2bf(hi) << 16);
}

// ---------------------------------------------------------------------------
// Kernel 0: pack W (f32, row-major ExD) into bf16 in WMMA-B lane layout.
// lane l of (n-tile ntg, k-chunk kc) holds expert n = ntg*16 + (l&15),
// K = kc*32 + (l>>4)*16 + [0..15] as 8 packed dwords (32B contiguous).
// Also zero-initializes the aux-loss accumulators each call (deterministic).
// ---------------------------------------------------------------------------
__global__ void wconv_kernel(const float* __restrict__ W,
                             unsigned int* __restrict__ wp,
                             float* __restrict__ cnt,
                             float* __restrict__ colsum) {
  const int tid = blockIdx.x * blockDim.x + threadIdx.x;   // 0 .. 65535
  if (blockIdx.x == 0 && threadIdx.x < E_EXP) {
    cnt[threadIdx.x]    = 0.0f;
    colsum[threadIdx.x] = 0.0f;
  }
  const int lane = tid & 31;
  const int kc   = (tid >> 5) & (NKC - 1);
  const int ntg  = tid >> 12;                               // 0..15
  const int n    = ntg * 16 + (lane & 15);
  const int kb   = kc * KC + ((lane >> 4) << 4);
  const float2* src = (const float2*)(W + (size_t)n * D_DIM + kb);
  unsigned int* dst = wp + (size_t)tid * 8;
#pragma unroll
  for (int h = 0; h < 8; ++h) {
    float2 p = src[h];
    dst[h] = pack2(p.x, p.y);
  }
}

// ---------------------------------------------------------------------------
// Kernel 1: fused GEMM (bf16 WMMA) + softmax + grouped top-k + aux partials.
// Block: 256 threads = 8 waves; 64 tokens x 256 experts per block.
// Wave w: mrow = w>>1 (16-token tile), nhalf = w&1 (128-expert half).
// x chunk staged in LDS as packed bf16 (double-buffered, single barrier/iter).
// ---------------------------------------------------------------------------
__global__ void __launch_bounds__(256, 1)
router_kernel(const float* __restrict__ x,
              const unsigned int* __restrict__ wp,
              float* __restrict__ out_w,
              int*   __restrict__ out_i,
              float* __restrict__ cnt,
              float* __restrict__ colsum) {
  __shared__ unsigned int xp[2][MB * AP_STR];  // 2 x 5 KB packed-bf16 x chunk
  __shared__ float lg[MB * E_EXP];             // 64 KB logits
  __shared__ float rowmax[MB];
  __shared__ float rowinv[MB];

  const int tid     = threadIdx.x;
  const int lane    = tid & 31;
  const int wave    = tid >> 5;
  const int mrow    = wave >> 1;          // 0..3
  const int nhalf   = wave & 1;           // 0..1
  const int khalf   = lane >> 4;          // 0..1
  const int mlane   = lane & 15;
  const int tokBase = blockIdx.x * MB;

  // staging role: each thread owns (row, seg): 8 floats -> 4 packed dwords
  const int srow = tid >> 2;              // 0..63
  const int sseg = tid & 3;               // 0..3
  const float4* __restrict__ gsrc =
      (const float4*)(x + (size_t)(tokBase + srow) * D_DIM) + sseg * 2;
  unsigned int* const sdst = &xp[0][0] + srow * AP_STR + sseg * 4;

  v8f acc[8] = {};                        // 8 tiles of 16x16 f32

  // this wave's packed-B stream (v8u = 32B per lane per tile-chunk)
  const v8u* __restrict__ wbase =
      (const v8u*)wp + ((size_t)(nhalf * 8) * NKC) * 32 + lane;

  // A gather base: two ds_load_b128 per chunk in the WMMA 16-bit A layout
  const unsigned int* const abase =
      &xp[0][0] + (mrow * 16 + mlane) * AP_STR + khalf * 8;

  float4 ra, rb;  // staged global data for next chunk

  // ---- prologue: fetch + convert + store chunk 0 into buffer 0 ----
  ra = gsrc[0];
  rb = gsrc[1];
  {
    uint4 pk;
    pk.x = pack2(ra.x, ra.y); pk.y = pack2(ra.z, ra.w);
    pk.z = pack2(rb.x, rb.y); pk.w = pack2(rb.z, rb.w);
    *(uint4*)sdst = pk;
  }
  __syncthreads();

  for (int kc = 0; kc < NKC; ++kc) {
    const bool more = (kc + 1 < NKC);
    // --- prefetch next chunk's x into registers (latency hidden by WMMA) ---
    if (more) {
      const float4* s = gsrc + (size_t)(kc + 1) * (KC / 4);
      ra = s[0];
      rb = s[1];
      __builtin_prefetch((const void*)(s + (KC / 4)), 0, 3);
    }

    // --- A: two ds_load_b128 from packed buffer (already WMMA layout) ---
    const unsigned int* ap = abase + (size_t)(kc & 1) * (MB * AP_STR);
    uint4 alo = *(const uint4*)(ap);       // dwords 0..3  (+khalf*8)
    uint4 ahi = *(const uint4*)(ap + 4);   // dwords 4..7  (+khalf*8)
    v8u araw;
    araw[0] = alo.x; araw[1] = alo.y; araw[2] = alo.z; araw[3] = alo.w;
    araw[4] = ahi.x; araw[5] = ahi.y; araw[6] = ahi.z; araw[7] = ahi.w;
    const v16bf A = __builtin_bit_cast(v16bf, araw);

    // --- 8 x (B load + WMMA), B pre-packed in lane layout (L2-resident) ---
    const v8u* wk = wbase + (size_t)kc * 32;
#pragma unroll
    for (int nt = 0; nt < 8; ++nt) {
      v8u braw = wk[(size_t)nt * (NKC * 32)];
      const v16bf B = __builtin_bit_cast(v16bf, braw);
      acc[nt] = __builtin_amdgcn_wmma_f32_16x16x32_bf16(
          false, A, false, B, (short)0, acc[nt], false, false);
    }

    // --- convert + store next chunk into the other buffer ---
    if (more) {
      uint4 pk;
      pk.x = pack2(ra.x, ra.y); pk.y = pack2(ra.z, ra.w);
      pk.z = pack2(rb.x, rb.y); pk.w = pack2(rb.z, rb.w);
      *(uint4*)(sdst + (size_t)((kc + 1) & 1) * (MB * AP_STR)) = pk;
    }
    __syncthreads();
  }

  // C layout: vgpr r, lane l -> (M = r + 8*(l>>4), N = l&15)
#pragma unroll
  for (int nt = 0; nt < 8; ++nt) {
#pragma unroll
    for (int r = 0; r < 8; ++r) {
      const int mm = mrow * 16 + r + 8 * khalf;
      const int nn = nhalf * 128 + nt * 16 + mlane;
      lg[mm * E_EXP + nn] = acc[nt][r];
    }
  }
  __syncthreads();

  // --- per-token routing: threads 0..63, one token each ---
  if (tid < MB) {
    const float* row = &lg[tid * E_EXP];
    float mx = -3.4e38f;
    for (int e = 0; e < E_EXP; ++e) mx = fmaxf(mx, row[e]);
    float s = 0.0f;
    for (int e = 0; e < E_EXP; ++e) s += exp2f((row[e] - mx) * LOG2E);
    const float inv = 1.0f / s;
    rowmax[tid] = mx;
    rowinv[tid] = inv;

    // top-2 logits per group (monotone under softmax -> same ids)
    float gv1[G_GRP], gv2[G_GRP], gs[G_GRP];
    int   gi1[G_GRP], gi2[G_GRP];
#pragma unroll
    for (int g = 0; g < G_GRP; ++g) {
      float v1 = -3.4e38f, v2 = -3.4e38f;
      int   i1 = 0,        i2 = 0;
      const float* gr = row + g * EPG;
      for (int e = 0; e < EPG; ++e) {
        const float v = gr[e];
        if (v > v1)      { v2 = v1; i2 = i1; v1 = v; i1 = e; }
        else if (v > v2) { v2 = v;  i2 = e; }
      }
      gv1[g] = v1; gv2[g] = v2; gi1[g] = i1; gi2[g] = i2;
      gs[g]  = exp2f((v1 - mx) * LOG2E) + exp2f((v2 - mx) * LOG2E);
    }
    // top-4 groups by score (descending, first-index on ties)
    int selg[NSEL];
    unsigned used = 0;
#pragma unroll
    for (int r = 0; r < NSEL; ++r) {
      float best = -3.4e38f; int bi = 0;
      for (int g = 0; g < G_GRP; ++g)
        if (!((used >> g) & 1u) && gs[g] > best) { best = gs[g]; bi = g; }
      used |= 1u << bi;
      selg[r] = bi;
    }
    int   ids[TOPK];
    float pw[TOPK];
    float psum = 0.0f;
#pragma unroll
    for (int r = 0; r < NSEL; ++r) {
      const int g = selg[r];
      ids[2 * r]     = g * EPG + gi1[g];
      ids[2 * r + 1] = g * EPG + gi2[g];
      pw[2 * r]      = exp2f((gv1[g] - mx) * LOG2E) * inv;
      pw[2 * r + 1]  = exp2f((gv2[g] - mx) * LOG2E) * inv;
      psum += pw[2 * r] + pw[2 * r + 1];
    }
    const float wn = 1.0f / (psum + 1e-9f);
    const int tok = tokBase + tid;
#pragma unroll
    for (int j = 0; j < TOPK; ++j) {
      out_w[(size_t)tok * TOPK + j] = pw[j] * wn;
      out_i[(size_t)tok * TOPK + j] = ids[j];
    }
    atomicAdd(&cnt[ids[0]], 1.0f);   // top1 usage histogram
  }
  __syncthreads();

  // --- per-expert prob column sums for aux loss (1 expert/thread) ---
  {
    const int e = tid;
    float a = 0.0f;
    for (int r = 0; r < MB; ++r) {
      a += exp2f((lg[r * E_EXP + e] - rowmax[r]) * LOG2E) * rowinv[r];
    }
    atomicAdd(&colsum[e], a);
  }
}

// ---------------------------------------------------------------------------
// Kernel 2: aux = E * sum_e (count_e/T) * (colsum_e/T)
// ---------------------------------------------------------------------------
__global__ void finalize_kernel(const float* __restrict__ cnt,
                                const float* __restrict__ colsum,
                                float* __restrict__ aux) {
  __shared__ float red[E_EXP];
  const int e = threadIdx.x;
  red[e] = cnt[e] * colsum[e];
  __syncthreads();
  for (int off = E_EXP / 2; off > 0; off >>= 1) {
    if (e < off) red[e] += red[e + off];
    __syncthreads();
  }
  if (e == 0) {
    const float t = (float)T_TOK;
    *aux = (float)E_EXP * red[0] / (t * t);
  }
}

// ---------------------------------------------------------------------------
extern "C" void kernel_launch(void* const* d_in, const int* in_sizes, int n_in,
                              void* d_out, int out_size, void* d_ws,
                              size_t ws_size, hipStream_t stream) {
  const float* x = (const float*)d_in[0];   // (T, D) f32
  const float* W = (const float*)d_in[1];   // (E, D) f32

  float* out_w = (float*)d_out;                                  // T*8 f32
  int*   out_i = (int*)((float*)d_out + (size_t)T_TOK * TOPK);   // T*8 i32
  float* aux   = (float*)d_out + (size_t)2 * T_TOK * TOPK;       // scalar

  // workspace: [0, 2MB) packed bf16 W ; then cnt[256], colsum[256]
  unsigned int* wp  = (unsigned int*)d_ws;
  float* cnt        = (float*)((char*)d_ws + (2u << 20));
  float* colsum     = cnt + E_EXP;

  wconv_kernel<<<256, 256, 0, stream>>>(W, wp, cnt, colsum);
  router_kernel<<<T_TOK / MB, 256, 0, stream>>>(x, wp, out_w, out_i, cnt,
                                                colsum);
  finalize_kernel<<<1, E_EXP, 0, stream>>>(cnt, colsum, aux);
}
